// GlobalPointer_54649163874412
// MI455X (gfx1250) — matrix operand order repaired and checked by reading, not compiled, not run on hardware
//
#include <hip/hip_runtime.h>

// ---------------------------------------------------------------------------
// MI455X (gfx1250, wave32) implementation of RoFormer attention logits.
//   K1: proj = x@W + b          (WMMA f16->f32, tiled through LDS)
//   K2: rotary embedding        (elementwise, writes q/k f16 head-major)
//   K3: logits = q@k^T, masked  (WMMA f16->f32, async-staged LDS tiles)
// Output-store bound (~100 MB fp32); all intermediates f16 and L2-resident.
// ---------------------------------------------------------------------------

typedef _Float16 v16h __attribute__((ext_vector_type(16)));
typedef _Float16 v8h  __attribute__((ext_vector_type(8)));
typedef _Float16 v2h  __attribute__((ext_vector_type(2)));
typedef float    v8f  __attribute__((ext_vector_type(8)));
typedef float    v4f  __attribute__((ext_vector_type(4)));
typedef int      v4i  __attribute__((ext_vector_type(4)));

#define NHEADS 12
#define HSZ    64
#define BB     8
#define SS     512
#define HH     1024
#define NOUT   1536            // NHEADS * HSZ * 2
#define BIGNEG 1000000000000.0f

// ---- CDNA5 async global->LDS staging (ASYNCcnt path), feature-guarded ------
#if __has_builtin(__builtin_amdgcn_global_load_async_to_lds_b128)
#define HAVE_ASYNC_LDS 1
typedef __attribute__((address_space(1))) v4i gv4i;   // global (AS1) int4
typedef __attribute__((address_space(3))) v4i lv4i;   // LDS    (AS3) int4
__device__ __forceinline__ void async_cp16(const void* g, void* l) {
  __builtin_amdgcn_global_load_async_to_lds_b128((gv4i*)g, (lv4i*)l, 0, 0);
}
__device__ __forceinline__ void async_wait0() {
#if __has_builtin(__builtin_amdgcn_s_wait_asynccnt)
  __builtin_amdgcn_s_wait_asynccnt(0);
#else
  asm volatile("s_wait_asynccnt 0x0" ::: "memory");
#endif
}
#endif

// ---- WMMA fragment builders (wave32 layouts per CDNA5 ISA 7.12.2) ----------
// A 16x32 f16: lane holds row (lane&15); halves [0..7] = K[lo..lo+7],
// halves [8..15] = K[16+lo..16+lo+7], lo = (lane<16 ? 0 : 8).
__device__ __forceinline__ v16h a_frag(const _Float16* row, int lane) {
  const int lo = (lane & 16) ? 8 : 0;
  v8h x = *(const v8h*)(row + lo);
  v8h y = *(const v8h*)(row + 16 + lo);
  return __builtin_shufflevector(x, y, 0,1,2,3,4,5,6,7,8,9,10,11,12,13,14,15);
}
// B 32x16 f16: lane holds column (lane&15); halves [0..7] = K[lo..lo+7],
// halves [8..15] = K[16+lo..16+lo+7], lo = (lane<16 ? 0 : 8).  `col` points at the K-major column (B^T row).
__device__ __forceinline__ v16h b_frag(const _Float16* col, int lane) {
  const int lo = (lane & 16) ? 8 : 0;
  v8h x = *(const v8h*)(col + lo);
  v8h y = *(const v8h*)(col + 16 + lo);
  return __builtin_shufflevector(x, y, 0,1,2,3,4,5,6,7,8,9,10,11,12,13,14,15);
}

// ---------------------------------------------------------------------------
// Kernel 1: proj[4096,1536] = x[4096,1024] @ W[1024,1536] + bias, stored f16.
// 64x64 block per WG (256 thr = 8 waves, 4(M) x 2(N)), wave = 16x32, K-step 32.
// W staged transposed into LDS with packed v2h (ds_store_b32) writes.
// ---------------------------------------------------------------------------
__global__ __launch_bounds__(256) void proj_gemm_kernel(
    const float* __restrict__ x, const float* __restrict__ W,
    const float* __restrict__ bias, _Float16* __restrict__ proj) {
  __shared__ __align__(32) _Float16 As[64 * 32];   // [m][k]
  __shared__ __align__(32) _Float16 Bs[64 * 32];   // [n][k]  (W transposed)
  const int tid  = threadIdx.x;
  const int lane = tid & 31;
  const int w    = tid >> 5;
  const int wm   = w & 3;            // 16-row strip
  const int wn   = w >> 2;           // 32-col strip
  const int mBase = blockIdx.x * 64;
  const int nBase = blockIdx.y * 64;

  v8f c0 = {0,0,0,0,0,0,0,0};
  v8f c1 = {0,0,0,0,0,0,0,0};

  const int ar = tid >> 2;           // 0..63  A stage row
  const int ac = (tid & 3) * 8;      // A stage col (8 floats)
  const int bk = (tid >> 4) * 2;     // 0..30  B stage k-pair (coalesced rows)
  const int bn = (tid & 15) * 4;     // 0..60  B stage n (4 floats)

  for (int k0 = 0; k0 < HH; k0 += 32) {
    // stage A tile, f32 -> f16
    const float* ap = x + (size_t)(mBase + ar) * HH + k0 + ac;
    v4f a0 = *(const v4f*)ap;
    v4f a1 = *(const v4f*)(ap + 4);
    v8h ah;
    ah[0]=(_Float16)a0[0]; ah[1]=(_Float16)a0[1]; ah[2]=(_Float16)a0[2]; ah[3]=(_Float16)a0[3];
    ah[4]=(_Float16)a1[0]; ah[5]=(_Float16)a1[1]; ah[6]=(_Float16)a1[2]; ah[7]=(_Float16)a1[3];
    *(v8h*)(As + ar * 32 + ac) = ah;
    // stage B tile: two consecutive W rows (k, k+1), 4 cols each; pack the
    // (k,k+1) halves of a column into one 4B LDS store (transposed layout).
    const float* bp0 = W + (size_t)(k0 + bk) * NOUT + nBase + bn;
    const float* bp1 = bp0 + NOUT;
    v4f b0 = *(const v4f*)bp0;
    v4f b1 = *(const v4f*)bp1;
#pragma unroll
    for (int j = 0; j < 4; ++j) {
      v2h p;
      p[0] = (_Float16)b0[j];
      p[1] = (_Float16)b1[j];
      *(v2h*)(Bs + (bn + j) * 32 + bk) = p;
    }
    __syncthreads();

    v16h a   = a_frag(As + (wm * 16 + (lane & 15)) * 32, lane);
    v16h bb0 = b_frag(Bs + (wn * 32 +      (lane & 15)) * 32, lane);
    v16h bb1 = b_frag(Bs + (wn * 32 + 16 + (lane & 15)) * 32, lane);
    c0 = __builtin_amdgcn_wmma_f32_16x16x32_f16(false, a, false, bb0, (short)0, c0, false, false);
    c1 = __builtin_amdgcn_wmma_f32_16x16x32_f16(false, a, false, bb1, (short)0, c1, false, false);
    __syncthreads();
  }

  // epilogue: bias + f16 store.  C layout: lane=col, VGPR r -> row r+8*(lane>=16)
  const int m0 = mBase + wm * 16 + (lane >> 4) * 8;
  const int n0 = nBase + wn * 32 + (lane & 15);
  const float bia0 = bias[n0];
  const float bia1 = bias[n0 + 16];
#pragma unroll
  for (int r = 0; r < 8; ++r) {
    const size_t rowoff = (size_t)(m0 + r) * NOUT;
    proj[rowoff + n0]      = (_Float16)(c0[r] + bia0);
    proj[rowoff + n0 + 16] = (_Float16)(c1[r] + bia1);
  }
}

// ---------------------------------------------------------------------------
// Kernel 2: rotary embedding.  One thread per (row, column-pair).
// theta index for output dim d is (d % 32) per the reference's tiled concat.
// q/k written head-major: [(b*12+h)*512 + s][64] f16.
// ---------------------------------------------------------------------------
__global__ __launch_bounds__(256) void rope_kernel(
    const _Float16* __restrict__ proj,
    _Float16* __restrict__ qb, _Float16* __restrict__ kb) {
  const int idx = blockIdx.x * 256 + threadIdx.x;   // 4096*768 threads exactly
  const int row = idx / 768;
  const int n   = (idx - row * 768) * 2;            // even column in [0,1536)
  const int h   = n >> 7;
  const int r   = n & 127;
  const int isk = r >> 6;
  const int d   = r & 63;                           // even
  const int s   = row & 511;
  const int b   = row >> 9;

  const float t0 = (float)proj[(size_t)row * NOUT + n];
  const float t1 = (float)proj[(size_t)row * NOUT + n + 1];

  const int   j0 = d & 31;                          // theta index = d % 32
  const float LN1E4_32 = 0.28782313662425f;         // ln(10000)/32
  const float th0 = __expf(-(float)j0       * LN1E4_32);
  const float th1 = __expf(-(float)(j0 + 1) * LN1E4_32);
  float s0, c0, s1, c1;
  __sincosf((float)s * th0, &s0, &c0);
  __sincosf((float)s * th1, &s1, &c1);

  const float o0 = t0 * c0 - t1 * s0;               // t*cos + (-t[d+1])*sin
  const float o1 = t1 * c1 + t0 * s1;               // t*cos + ( t[d]  )*sin

  _Float16* dst = isk ? kb : qb;
  const size_t o = (((size_t)(b * NHEADS + h)) * SS + s) * HSZ + d;
  dst[o]     = (_Float16)o0;
  dst[o + 1] = (_Float16)o1;
}

// ---------------------------------------------------------------------------
// Kernel 3: logits[b,h,m,n] = (masked) q·k^T / 8.
// Per WG: 64(m) x 128(n) block of one head; 8 waves (4 M-strips x 2 N-strips),
// each wave 16x64 via 4 accumulators, K=64 in two 32-steps.
// Tiles staged with GLOBAL_LOAD_ASYNC_TO_LDS_B128 when available.
// ---------------------------------------------------------------------------
__global__ __launch_bounds__(256) void attn_logits_kernel(
    const _Float16* __restrict__ qb, const _Float16* __restrict__ kb,
    const float* __restrict__ mask, float* __restrict__ out) {
  __shared__ __align__(32) _Float16 Qs[64 * 64];
  __shared__ __align__(32) _Float16 Ks[128 * 64];
  const int tid  = threadIdx.x;
  const int lane = tid & 31;
  const int w    = tid >> 5;
  const int wm   = w & 3;
  const int wn   = w >> 2;
  const int mBase = blockIdx.x * 64;
  const int nBase = blockIdx.y * 128;
  const int z     = blockIdx.z;          // b*12 + h
  const int b     = z / NHEADS;

  const _Float16* qh = qb + (size_t)z * SS * HSZ;
  const _Float16* kh = kb + (size_t)z * SS * HSZ;

  {  // stage Q (64x64) and K (128x64) tiles
    const int rr = tid >> 2;             // 0..63
    const int cc = (tid & 3) * 16;       // 0,16,32,48
#ifdef HAVE_ASYNC_LDS
    async_cp16(qh + (size_t)(mBase + rr) * HSZ + cc,     Qs + rr * 64 + cc);
    async_cp16(qh + (size_t)(mBase + rr) * HSZ + cc + 8, Qs + rr * 64 + cc + 8);
#pragma unroll
    for (int i = 0; i < 2; ++i) {
      const int kr = i * 64 + rr;
      async_cp16(kh + (size_t)(nBase + kr) * HSZ + cc,     Ks + kr * 64 + cc);
      async_cp16(kh + (size_t)(nBase + kr) * HSZ + cc + 8, Ks + kr * 64 + cc + 8);
    }
    async_wait0();
#else
    *(v8h*)(Qs + rr * 64 + cc)     = *(const v8h*)(qh + (size_t)(mBase + rr) * HSZ + cc);
    *(v8h*)(Qs + rr * 64 + cc + 8) = *(const v8h*)(qh + (size_t)(mBase + rr) * HSZ + cc + 8);
#pragma unroll
    for (int i = 0; i < 2; ++i) {
      const int kr = i * 64 + rr;
      *(v8h*)(Ks + kr * 64 + cc)     = *(const v8h*)(kh + (size_t)(nBase + kr) * HSZ + cc);
      *(v8h*)(Ks + kr * 64 + cc + 8) = *(const v8h*)(kh + (size_t)(nBase + kr) * HSZ + cc + 8);
    }
#endif
  }
  __syncthreads();

  v8f acc[4];
#pragma unroll
  for (int j = 0; j < 4; ++j) acc[j] = (v8f){0,0,0,0,0,0,0,0};

#pragma unroll
  for (int ks = 0; ks < 2; ++ks) {
    v16h a = a_frag(Qs + (wm * 16 + (lane & 15)) * 64 + ks * 32, lane);
#pragma unroll
    for (int j = 0; j < 4; ++j) {
      v16h bbj = b_frag(Ks + (wn * 64 + j * 16 + (lane & 15)) * 64 + ks * 32, lane);
      acc[j] = __builtin_amdgcn_wmma_f32_16x16x32_f16(false, a, false, bbj, (short)0, acc[j], false, false);
    }
  }

  // epilogue: row mask, col mask, strict-lower-tri -1e12, then * 1/8
  const int m0 = mBase + wm * 16 + (lane >> 4) * 8;
#pragma unroll
  for (int j = 0; j < 4; ++j) {
    const int n  = nBase + wn * 64 + j * 16 + (lane & 15);
    const float mc = mask[b * SS + n];
#pragma unroll
    for (int r = 0; r < 8; ++r) {
      const int m  = m0 + r;
      const float mr = mask[b * SS + m];
      float v = acc[j][r];
      v = v * mr - BIGNEG * (1.0f - mr);
      v = v * mc - BIGNEG * (1.0f - mc);
      if (m > n) v -= BIGNEG;
      out[((size_t)z * SS + m) * SS + n] = v * 0.125f;
    }
  }
}

// ---------------------------------------------------------------------------
extern "C" void kernel_launch(void* const* d_in, const int* in_sizes, int n_in,
                              void* d_out, int out_size, void* d_ws, size_t ws_size,
                              hipStream_t stream) {
  const float* x    = (const float*)d_in[0];   // (8,512,1024)
  const float* mask = (const float*)d_in[1];   // (8,512)
  const float* W    = (const float*)d_in[2];   // (1024,1536)
  const float* bias = (const float*)d_in[3];   // (1536,)
  float* out = (float*)d_out;                  // (8,12,512,512)

  _Float16* proj = (_Float16*)d_ws;                          // 4096*1536 f16
  _Float16* qb   = proj + (size_t)4096 * NOUT;               // 96*512*64 f16
  _Float16* kb   = qb   + (size_t)BB * NHEADS * SS * HSZ;

  proj_gemm_kernel<<<dim3(4096 / 64, NOUT / 64), 256, 0, stream>>>(x, W, bias, proj);
  rope_kernel<<<dim3((4096 * 768) / 256), 256, 0, stream>>>(proj, qb, kb);
  attn_logits_kernel<<<dim3(SS / 64, SS / 128, BB * NHEADS), 256, 0, stream>>>(qb, kb, mask, out);
}
